// Block_44040594653419
// MI455X (gfx1250) — matrix operand order repaired and checked by reading, not compiled, not run on hardware
//
#include <hip/hip_runtime.h>
#include <hip/hip_bf16.h>
#include <math.h>
#include <stdint.h>

typedef __attribute__((ext_vector_type(16))) _Float16 v16h;
typedef __attribute__((ext_vector_type(8)))  float    v8f;
typedef _Float16 half_t;

#define DIMV 1024
#define NHV  16
#define HDV  64
#define BBV  4
#define SSV  2048
#define NTOK (BBV*SSV)   /* 8192 tokens */
#define EPSV 1e-5f

union FragH { unsigned u[8]; v16h h; };

__device__ __forceinline__ float gelu_exact(float v) {
    return 0.5f * v * (1.0f + erff(v * 0.70710678118654752440f));
}

// ---- CDNA5 async global->LDS copy (ASYNCcnt-tracked) ----
__device__ __forceinline__ void async_b128(unsigned lds_off, const half_t* g) {
    asm volatile("global_load_async_to_lds_b128 %0, %1, off"
                 :: "v"(lds_off), "v"((unsigned long long)(uintptr_t)g)
                 : "memory");
}
__device__ __forceinline__ void wait_async0() {
    asm volatile("s_wait_asynccnt 0x0" ::: "memory");
}

__device__ __forceinline__ float block_sum256(float v, float* red) {
    int t = threadIdx.x;
    red[t] = v; __syncthreads();
    for (int off = 128; off > 0; off >>= 1) {
        if (t < off) red[t] += red[t + off];
        __syncthreads();
    }
    float r = red[0];
    __syncthreads();
    return r;
}

// ---------------- weight packing ----------------

__global__ void pack_qkv_kernel(const float* __restrict__ Wq, const float* __restrict__ Wk,
                                const float* __restrict__ Wv,
                                const float* __restrict__ bq, const float* __restrict__ bk,
                                const float* __restrict__ bv,
                                half_t* __restrict__ Wt, float* __restrict__ biasq) {
    size_t i = (size_t)blockIdx.x * 256 + threadIdx.x;   // over 3072*1024
    if (i >= (size_t)3072 * 1024) return;
    int c = (int)(i >> 10);          // packed row = logical output column 0..3071
    int d = (int)(i & 1023);         // K index
    int sec  = c >> 10;
    int head = (c & 1023) >> 6;
    int kk   = c & 63;
    const float* W = (sec == 0) ? Wq : (sec == 1) ? Wk : Wv;
    Wt[i] = (half_t)W[((size_t)head * DIMV + d) * HDV + kk];
    if (d == 0) {
        const float* bb = (sec == 0) ? bq : (sec == 1) ? bk : bv;
        biasq[c] = bb[head * HDV + kk];
    }
}

__global__ void pack_w1_kernel(const float* __restrict__ W1, half_t* __restrict__ W1t) {
    size_t i = (size_t)blockIdx.x * 256 + threadIdx.x;   // over 4096*1024, [n][d]
    if (i >= (size_t)4096 * 1024) return;
    int n = (int)(i >> 10);
    int d = (int)(i & 1023);
    W1t[i] = (half_t)W1[(size_t)d * 4096 + n];
}

__global__ void pack_w2_kernel(const float* __restrict__ W2, half_t* __restrict__ W2t) {
    size_t i = (size_t)blockIdx.x * 256 + threadIdx.x;   // over 1024*4096, [n][d]
    if (i >= (size_t)1024 * 4096) return;
    int n = (int)(i >> 12);
    int d = (int)(i & 4095);
    W2t[i] = (half_t)W2[(size_t)d * 1024 + n];
}

// ---------------- layer norms ----------------

__global__ __launch_bounds__(256)
void ln1_kernel(const float* __restrict__ x, const float* __restrict__ g,
                const float* __restrict__ b, half_t* __restrict__ out) {
    __shared__ float red[256];
    int row = blockIdx.x;
    const float* xr = x + (size_t)row * DIMV;
    float v[4]; float s = 0.f;
    #pragma unroll
    for (int i = 0; i < 4; i++) { v[i] = xr[threadIdx.x + i * 256]; s += v[i]; }
    float mu = block_sum256(s, red) * (1.0f / DIMV);
    float s2 = 0.f;
    #pragma unroll
    for (int i = 0; i < 4; i++) { float d = v[i] - mu; s2 += d * d; }
    float rs = rsqrtf(block_sum256(s2, red) * (1.0f / DIMV) + EPSV);
    #pragma unroll
    for (int i = 0; i < 4; i++) {
        int c = threadIdx.x + i * 256;
        out[(size_t)row * DIMV + c] = (half_t)((v[i] - mu) * rs * g[c] + b[c]);
    }
}

__global__ __launch_bounds__(256)
void double_ln_kernel(const float* __restrict__ x,
                      const float* __restrict__ g1, const float* __restrict__ b1,
                      const float* __restrict__ g2, const float* __restrict__ b2,
                      half_t* __restrict__ out) {
    __shared__ float red[256];
    int row = blockIdx.x;
    const float* xr = x + (size_t)row * DIMV;
    float v[4]; float s = 0.f;
    #pragma unroll
    for (int i = 0; i < 4; i++) { v[i] = xr[threadIdx.x + i * 256]; s += v[i]; }
    float mu = block_sum256(s, red) * (1.0f / DIMV);
    float s2 = 0.f;
    #pragma unroll
    for (int i = 0; i < 4; i++) { float d = v[i] - mu; s2 += d * d; }
    float rs = rsqrtf(block_sum256(s2, red) * (1.0f / DIMV) + EPSV);
    float y[4]; float t = 0.f;
    #pragma unroll
    for (int i = 0; i < 4; i++) {
        int c = threadIdx.x + i * 256;
        y[i] = (v[i] - mu) * rs * g1[c] + b1[c];
        t += y[i];
    }
    float mu2 = block_sum256(t, red) * (1.0f / DIMV);
    float t2 = 0.f;
    #pragma unroll
    for (int i = 0; i < 4; i++) { float d = y[i] - mu2; t2 += d * d; }
    float rs2 = rsqrtf(block_sum256(t2, red) * (1.0f / DIMV) + EPSV);
    #pragma unroll
    for (int i = 0; i < 4; i++) {
        int c = threadIdx.x + i * 256;
        out[(size_t)row * DIMV + c] = (half_t)((y[i] - mu2) * rs2 * g2[c] + b2[c]);
    }
}

__global__ void add_kernel(const float* __restrict__ a, const float* __restrict__ b,
                           float* __restrict__ c, int n) {
    int i = blockIdx.x * 256 + threadIdx.x;
    if (i < n) c[i] = a[i] + b[i];
}

// ---------------- WMMA GEMM  (A [M,K] f16 row-major, Bt [N,K] f16 row-major) ----------------
// Block: 256 threads = 8 waves (4 in M x 2 in N), tile 128(M) x 128(N), K-step 32,
// double-buffered LDS fed by global_load_async_to_lds_b128.
// Per wave per K-step: 2 A-frags + 4 B-frags -> 8 WMMAs.
// EPI 0: QKV routing (bias + q scale fold, V stored transposed)
// EPI 1: bias + GELU -> f16 out [M,N]
// EPI 2: bias + GELU + residual -> f32 out [M,N]

template<int EPI>
__global__ __launch_bounds__(256)
void gemm_wmma_kernel(const half_t* __restrict__ A, const half_t* __restrict__ Bt,
                      const float* __restrict__ bias,
                      float* __restrict__ outF, half_t* __restrict__ outH,
                      const float* __restrict__ resid,
                      half_t* __restrict__ qbuf, half_t* __restrict__ kbuf,
                      half_t* __restrict__ vbuf,
                      int M, int N, int K) {
    __shared__ __align__(16) half_t ldsA[2][128 * 40];
    __shared__ __align__(16) half_t ldsB[2][128 * 40];
    const int tid    = threadIdx.x;
    const int lane   = tid & 31;
    const int wave   = tid >> 5;
    const int wave_m = wave & 3;
    const int wave_n = wave >> 2;
    const int hi     = (lane >> 4) & 1;
    const int ln     = lane & 15;
    const int m0     = blockIdx.y * 128;
    const int n0     = blockIdx.x * 128;

    v8f acc[2][4];
    #pragma unroll
    for (int f = 0; f < 2; f++)
        #pragma unroll
        for (int i = 0; i < 4; i++) acc[f][i] = 0.0f;

    auto issue_stage = [&](int buf, int k0) {
        #pragma unroll
        for (int vv = 0; vv < 2; vv++) {
            int idx  = tid * 2 + vv;        // 0..511
            int row  = idx >> 2;            // 4 x 16B per 32-half row
            int koff = (idx & 3) * 8;
            async_b128((unsigned)(uintptr_t)&ldsA[buf][row * 40 + koff],
                       A + (size_t)(m0 + row) * K + k0 + koff);
            async_b128((unsigned)(uintptr_t)&ldsB[buf][row * 40 + koff],
                       Bt + (size_t)(n0 + row) * K + k0 + koff);
        }
    };

    issue_stage(0, 0);
    for (int k0 = 0; k0 < K; k0 += 32) {
        const int buf = (k0 >> 5) & 1;
        wait_async0();            // this wave's stage `buf` writes are in LDS
        __syncthreads();          // everyone's writes visible; prior readers of buf^1 done
        if (k0 + 32 < K) issue_stage(buf ^ 1, k0 + 32);

        FragH a[2];
        #pragma unroll
        for (int f = 0; f < 2; f++) {
            const half_t* p = &ldsA[buf][(wave_m * 32 + f * 16 + ln) * 40];
            #pragma unroll
            for (int t = 0; t < 8; t++) {
                int kk = (t < 4 ? 2 * t : 2 * t + 8) + (hi ? 8 : 0);
                a[f].u[t] = *(const unsigned*)(p + kk);
            }
        }
        #pragma unroll
        for (int nf = 0; nf < 4; nf++) {
            FragH bfr;
            const half_t* p = &ldsB[buf][(wave_n * 64 + nf * 16 + ln) * 40];
            #pragma unroll
            for (int t = 0; t < 8; t++)
                bfr.u[t] = *(const unsigned*)(p + (t < 4 ? 2 * t : 2 * t + 8) + (hi ? 8 : 0));
            #pragma unroll
            for (int f = 0; f < 2; f++)
                acc[f][nf] = __builtin_amdgcn_wmma_f32_16x16x32_f16(
                    false, a[f].h, false, bfr.h, (short)0, acc[f][nf], false, false);
        }
    }

    // epilogue (C layout: VGPR r -> row r+8*hi, col = lane&15)
    #pragma unroll
    for (int f = 0; f < 2; f++) {
        #pragma unroll
        for (int nf = 0; nf < 4; nf++) {
            #pragma unroll
            for (int r = 0; r < 8; r++) {
                int row = m0 + wave_m * 32 + f * 16 + r + hi * 8;
                int col = n0 + wave_n * 64 + nf * 16 + ln;
                float val = acc[f][nf][r] + bias[col];
                if (EPI == 0) {
                    int bI   = row >> 11;         // /S (2048)
                    int sI   = row & 2047;
                    int sec  = col >> 10;
                    int head = (col & 1023) >> 6;
                    int kk   = col & 63;
                    if (sec == 0) {
                        qbuf[(((size_t)(bI * NHV + head) * SSV) + sI) * HDV + kk] =
                            (half_t)(val * 0.125f);   // fold 1/sqrt(64)
                    } else if (sec == 1) {
                        kbuf[(((size_t)(bI * NHV + head) * SSV) + sI) * HDV + kk] = (half_t)val;
                    } else {
                        vbuf[(((size_t)(bI * NHV + head) * HDV) + kk) * SSV + sI] = (half_t)val;
                    }
                } else if (EPI == 1) {
                    outH[(size_t)row * N + col] = (half_t)gelu_exact(val);
                } else {
                    outF[(size_t)row * N + col] =
                        resid[(size_t)row * N + col] + gelu_exact(val);
                }
            }
        }
    }
}

// ---------------- flash attention ----------------
// grid (S/64, NH, B), 128 threads = 4 waves; each wave owns 16 query rows.
// Key tile = 64, double-buffered K/V staged via async-to-LDS.
// q,k: [B,NH,S,HD] f16 (q pre-scaled by 1/8).  vT: [B,NH,HD,S] f16.

__global__ __launch_bounds__(128)
void flash_attn_kernel(const half_t* __restrict__ q, const half_t* __restrict__ k,
                       const half_t* __restrict__ vT, float* __restrict__ obuf) {
    __shared__ __align__(16) half_t ldsK[2][64 * 72];     // [key][hd]
    __shared__ __align__(16) half_t ldsV[2][64 * 72];     // [hd][key] (= V^T tile)
    __shared__ __align__(16) half_t ldsP[4][16 * 72];     // per-wave P staging
    const int tid  = threadIdx.x;
    const int lane = tid & 31;
    const int wave = tid >> 5;
    const int hi   = (lane >> 4) & 1;
    const int ln   = lane & 15;
    const int q0   = blockIdx.x * 64;
    const int h    = blockIdx.y;
    const int bI   = blockIdx.z;
    const size_t bh = (size_t)(bI * NHV + h);
    const half_t* qp = q  + bh * SSV * HDV;
    const half_t* kp = k  + bh * SSV * HDV;
    const half_t* vp = vT + bh * HDV * SSV;

    // Q fragments (A layout), 2 k-chunks over hd=64
    FragH qa[2];
    {
        const half_t* qr = qp + (size_t)(q0 + wave * 16 + ln) * HDV;
        #pragma unroll
        for (int c = 0; c < 2; c++)
            #pragma unroll
            for (int t = 0; t < 8; t++) {
                int kk = (t < 4 ? 2 * t : 2 * t + 8) + (hi ? 8 : 0) + c * 32;
                qa[c].u[t] = *(const unsigned*)(qr + kk);
            }
    }

    float mrow[8], lrow[8];
    v8f o[4];
    #pragma unroll
    for (int r = 0; r < 8; r++) { mrow[r] = -3.0e38f; lrow[r] = 0.f; }
    #pragma unroll
    for (int i = 0; i < 4; i++) o[i] = 0.0f;

    auto issue_tile = [&](int buf, int kb) {
        #pragma unroll
        for (int vv = 0; vv < 4; vv++) {
            int idx  = tid * 4 + vv;       // 0..511
            int row  = idx >> 3;           // 8 x 16B per 64-half row
            int koff = (idx & 7) * 8;
            async_b128((unsigned)(uintptr_t)&ldsK[buf][row * 72 + koff],
                       kp + (size_t)(kb + row) * HDV + koff);
            async_b128((unsigned)(uintptr_t)&ldsV[buf][row * 72 + koff],
                       vp + (size_t)row * SSV + kb + koff);
        }
    };

    const int ntiles = blockIdx.x + 1;     // keys 0 .. q0+63, 64 per tile
    issue_tile(0, 0);
    for (int kt = 0; kt < ntiles; kt++) {
        const int kb  = kt * 64;
        const int buf = kt & 1;
        wait_async0();
        __syncthreads();
        if (kt + 1 < ntiles) issue_tile(buf ^ 1, kb + 64);

        // scores: S = Q * K^T  (N = 64 local keys, K-dim = hd 64)
        v8f sc[4];
        #pragma unroll
        for (int i = 0; i < 4; i++) sc[i] = 0.0f;
        #pragma unroll
        for (int nf = 0; nf < 4; nf++) {
            #pragma unroll
            for (int c = 0; c < 2; c++) {
                FragH bfr;
                const half_t* p = &ldsK[buf][(nf * 16 + ln) * 72 + c * 32];
                #pragma unroll
                for (int t = 0; t < 8; t++)
                    bfr.u[t] = *(const unsigned*)(p + (t < 4 ? 2 * t : 2 * t + 8) + (hi ? 8 : 0));
                sc[nf] = __builtin_amdgcn_wmma_f32_16x16x32_f16(
                    false, qa[c].h, false, bfr.h, (short)0, sc[nf], false, false);
            }
        }

        // online softmax (mask only on the diagonal tile)
        const int qbase = q0 + wave * 16 + hi * 8;
        const bool lastTile = (kt == ntiles - 1);
        half_t* pw = ldsP[wave];
        #pragma unroll
        for (int r = 0; r < 8; r++) {
            int qi = qbase + r;
            float s0 = sc[0][r], s1 = sc[1][r], s2 = sc[2][r], s3 = sc[3][r];
            if (lastTile) {
                if (kb + ln      > qi) s0 = -1e9f;
                if (kb + 16 + ln > qi) s1 = -1e9f;
                if (kb + 32 + ln > qi) s2 = -1e9f;
                if (kb + 48 + ln > qi) s3 = -1e9f;
            }
            float vmx = fmaxf(fmaxf(s0, s1), fmaxf(s2, s3));
            #pragma unroll
            for (int off = 8; off > 0; off >>= 1)
                vmx = fmaxf(vmx, __shfl_xor(vmx, off, 16));
            float mnew = fmaxf(mrow[r], vmx);
            float scal = __expf(mrow[r] - mnew);
            float e0 = __expf(s0 - mnew);
            float e1 = __expf(s1 - mnew);
            float e2 = __expf(s2 - mnew);
            float e3 = __expf(s3 - mnew);
            float sum = (e0 + e1) + (e2 + e3);
            #pragma unroll
            for (int off = 8; off > 0; off >>= 1)
                sum += __shfl_xor(sum, off, 16);
            lrow[r] = lrow[r] * scal + sum;
            mrow[r] = mnew;
            int row = r + hi * 8;
            pw[row * 72 + ln]      = (half_t)e0;
            pw[row * 72 + 16 + ln] = (half_t)e1;
            pw[row * 72 + 32 + ln] = (half_t)e2;
            pw[row * 72 + 48 + ln] = (half_t)e3;
            #pragma unroll
            for (int nf = 0; nf < 4; nf++) o[nf][r] *= scal;
        }
        __syncthreads();

        // O += P * V   (N = hd 64, K-dim = 64 keys)
        FragH pa[2];
        #pragma unroll
        for (int c = 0; c < 2; c++)
            #pragma unroll
            for (int t = 0; t < 8; t++) {
                int kk = (t < 4 ? 2 * t : 2 * t + 8) + (hi ? 8 : 0) + c * 32;
                pa[c].u[t] = *(const unsigned*)(pw + ln * 72 + kk);
            }
        #pragma unroll
        for (int nf = 0; nf < 4; nf++) {
            #pragma unroll
            for (int c = 0; c < 2; c++) {
                FragH bfr;
                const half_t* p = &ldsV[buf][(nf * 16 + ln) * 72 + c * 32];
                #pragma unroll
                for (int t = 0; t < 8; t++)
                    bfr.u[t] = *(const unsigned*)(p + (t < 4 ? 2 * t : 2 * t + 8) + (hi ? 8 : 0));
                o[nf] = __builtin_amdgcn_wmma_f32_16x16x32_f16(
                    false, pa[c].h, false, bfr.h, (short)0, o[nf], false, false);
            }
        }
    }

    // write O / l, already laid out as concat-heads [B,S,DIM]
    #pragma unroll
    for (int nf = 0; nf < 4; nf++) {
        #pragma unroll
        for (int r = 0; r < 8; r++) {
            int qi = q0 + wave * 16 + r + hi * 8;
            int hd = nf * 16 + ln;
            obuf[((size_t)(bI * SSV + qi)) * DIMV + h * HDV + hd] = o[nf][r] / lrow[r];
        }
    }
}

// ---------------- launcher ----------------

extern "C" void kernel_launch(void* const* d_in, const int* in_sizes, int n_in,
                              void* d_out, int out_size, void* d_ws, size_t ws_size,
                              hipStream_t stream) {
    (void)in_sizes; (void)n_in; (void)out_size; (void)ws_size;
    const float* x      = (const float*)d_in[0];
    const float* ln1_g  = (const float*)d_in[1];
    const float* ln1_b  = (const float*)d_in[2];
    const float* Wq     = (const float*)d_in[3];
    const float* bq     = (const float*)d_in[4];
    const float* Wk     = (const float*)d_in[5];
    const float* bk     = (const float*)d_in[6];
    const float* Wv     = (const float*)d_in[7];
    const float* bv     = (const float*)d_in[8];
    const float* ln2_g  = (const float*)d_in[9];
    const float* ln2_b  = (const float*)d_in[10];
    const float* fcln_g = (const float*)d_in[11];
    const float* fcln_b = (const float*)d_in[12];
    const float* W1     = (const float*)d_in[13];
    const float* b1     = (const float*)d_in[14];
    const float* W2     = (const float*)d_in[15];
    const float* b2     = (const float*)d_in[16];
    float* out = (float*)d_out;

    char* ws = (char*)d_ws;
    size_t off = 0;
    auto alloc = [&](size_t bytes) -> void* {
        void* p = ws + off;
        off += (bytes + 255) & ~(size_t)255;
        return p;
    };
    half_t* hf16    = (half_t*)alloc((size_t)NTOK * DIMV * 2);
    half_t* Wqkvt   = (half_t*)alloc((size_t)3072 * 1024 * 2);
    float*  biasqkv = (float*) alloc((size_t)3072 * 4);
    half_t* qb      = (half_t*)alloc((size_t)BBV * NHV * SSV * HDV * 2);
    half_t* kb      = (half_t*)alloc((size_t)BBV * NHV * SSV * HDV * 2);
    half_t* vb      = (half_t*)alloc((size_t)BBV * NHV * SSV * HDV * 2);
    float*  obuf    = (float*) alloc((size_t)NTOK * DIMV * 4);
    float*  x2      = (float*) alloc((size_t)NTOK * DIMV * 4);
    half_t* h2      = (half_t*)alloc((size_t)NTOK * DIMV * 2);
    half_t* W1t     = (half_t*)alloc((size_t)4096 * 1024 * 2);
    half_t* W2t     = (half_t*)alloc((size_t)1024 * 4096 * 2);
    half_t* mid     = (half_t*)alloc((size_t)NTOK * 4096 * 2);

    // weight prep
    pack_qkv_kernel<<<(3072 * 1024 + 255) / 256, 256, 0, stream>>>(
        Wq, Wk, Wv, bq, bk, bv, Wqkvt, biasqkv);
    pack_w1_kernel<<<(4096 * 1024 + 255) / 256, 256, 0, stream>>>(W1, W1t);
    pack_w2_kernel<<<(1024 * 4096 + 255) / 256, 256, 0, stream>>>(W2, W2t);

    // LN1 -> f16
    ln1_kernel<<<NTOK, 256, 0, stream>>>(x, ln1_g, ln1_b, hf16);

    // fused QKV GEMM: [8192,1024] x [1024,3072]
    gemm_wmma_kernel<0><<<dim3(3072 / 128, NTOK / 128), 256, 0, stream>>>(
        hf16, Wqkvt, biasqkv, nullptr, nullptr, nullptr, qb, kb, vb,
        NTOK, 3072, DIMV);

    // flash attention
    flash_attn_kernel<<<dim3(SSV / 64, NHV, BBV), 128, 0, stream>>>(qb, kb, vb, obuf);

    // x2 = x + attn_out
    add_kernel<<<(NTOK * DIMV + 255) / 256, 256, 0, stream>>>(x, obuf, x2, NTOK * DIMV);

    // double layernorm -> f16
    double_ln_kernel<<<NTOK, 256, 0, stream>>>(x2, ln2_g, ln2_b, fcln_g, fcln_b, h2);

    // FC1: [8192,1024] x [1024,4096] + b1, GELU -> f16
    gemm_wmma_kernel<1><<<dim3(4096 / 128, NTOK / 128), 256, 0, stream>>>(
        h2, W1t, b1, nullptr, mid, nullptr, nullptr, nullptr, nullptr,
        NTOK, 4096, DIMV);

    // FC2: [8192,4096] x [4096,1024] + b2, GELU, + x2 -> f32 out
    gemm_wmma_kernel<2><<<dim3(1024 / 128, NTOK / 128), 256, 0, stream>>>(
        mid, W2t, b2, out, nullptr, x2, nullptr, nullptr, nullptr,
        NTOK, DIMV, 4096);
}
